// MultiPositiveInfoNCELoss_46016279610196
// MI455X (gfx1250) — matrix operand order repaired and checked by reading, not compiled, not run on hardware
//
#include <hip/hip_runtime.h>

typedef __attribute__((ext_vector_type(16))) _Float16 v16h;
typedef __attribute__((ext_vector_type(8)))  _Float16 v8h;
typedef __attribute__((ext_vector_type(4)))  _Float16 v4h;
typedef __attribute__((ext_vector_type(8)))  float    v8f;

#define BB   4096      // images
#define VV   5         // texts per image
#define DD   512       // feature dim
#define BVV  20480     // total texts
#define INVT 14.285714285714286f   // 1/0.07, also the constant LSE shift (max |logit|)

#define BM 128
#define BN 128
#define LDA 40         // 32 k-values + 8 f16 pad (16B-aligned rows, conflict-free b128 access)

// ---- WMMA fragment loads from row-major LDS tiles (ISA 7.12.2 layouts) ----

// A 16x32 f16: lane<16 -> row=lane, K {0..7, 16..23}; lane>=16 -> row=lane-16, K {8..15, 24..31}
__device__ __forceinline__ v16h load_frag_a(const _Float16* base, int lane) {
  int row = lane & 15;
  int kh  = (lane >> 4) << 3;
  v8h x0 = *reinterpret_cast<const v8h*>(base + row * LDA + kh);
  v8h x1 = *reinterpret_cast<const v8h*>(base + row * LDA + kh + 16);
  return __builtin_shufflevector(x0, x1, 0,1,2,3,4,5,6,7,8,9,10,11,12,13,14,15);
}

// B 32x16 f16 (from row-major [col][k] tile): lane<16 -> col=lane, K {0..7, 16..23}; lane>=16 -> col=lane-16, K {8..15, 24..31}
__device__ __forceinline__ v16h load_frag_b(const _Float16* base, int lane) {
  int col = lane & 15;
  int kb  = (lane >> 4) << 3;
  v8h x0 = *reinterpret_cast<const v8h*>(base + col * LDA + kb);
  v8h x1 = *reinterpret_cast<const v8h*>(base + col * LDA + kb + 16);
  return __builtin_shufflevector(x0, x1, 0,1,2,3,4,5,6,7,8,9,10,11,12,13,14,15);
}

// ---------------- Kernel 0: one-time f32 -> (hi,lo) f16 split ----------------
__global__ __launch_bounds__(256)
void split_f16(const float* __restrict__ src, _Float16* __restrict__ hi,
               _Float16* __restrict__ lo, int n4) {
  int i = blockIdx.x * 256 + threadIdx.x;     // float4 index
  if (i >= n4) return;
  float4 v = reinterpret_cast<const float4*>(src)[i];
  v4h H, L;
  #pragma unroll
  for (int e = 0; e < 4; ++e) {
    float x = (&v.x)[e];
    _Float16 h = (_Float16)x;
    H[e] = h;
    L[e] = (_Float16)(x - (float)h);
  }
  reinterpret_cast<v4h*>(hi)[i] = H;
  reinterpret_cast<v4h*>(lo)[i] = L;
}

// ---------------- Kernel 1: fused GEMM + exp + partial row/col sums ----------------
__global__ __launch_bounds__(256)
void gemm_lse_partials(const _Float16* __restrict__ imgH, const _Float16* __restrict__ imgL,
                       const _Float16* __restrict__ txtH, const _Float16* __restrict__ txtL,
                       float* __restrict__ rowG,   // [160][4096]
                       float* __restrict__ colG) { // [32][20480]
  __shared__ _Float16 sAh[2][BM * LDA];
  __shared__ _Float16 sAl[2][BM * LDA];
  __shared__ _Float16 sBh[2][BN * LDA];
  __shared__ _Float16 sBl[2][BN * LDA];
  __shared__ float rowPart[8][64];
  __shared__ float colPart[8][32];

  const int tid  = threadIdx.x;
  const int lane = tid & 31;
  const int w    = tid >> 5;   // wave 0..7
  const int wr   = w >> 2;     // wave row 0..1 (64 rows each)
  const int wc   = w & 3;      // wave col 0..3 (32 cols each)
  const int bx   = blockIdx.x; // column tile 0..159
  const int by   = blockIdx.y; // row tile    0..31

  // uint4 = 8 f16; each row of a matrix is 64 uint4s (512 f16).
  const uint4* pAh = reinterpret_cast<const uint4*>(imgH);
  const uint4* pAl = reinterpret_cast<const uint4*>(imgL);
  const uint4* pBh = reinterpret_cast<const uint4*>(txtH);
  const uint4* pBl = reinterpret_cast<const uint4*>(txtL);

  v8f acc[4][2];
  {
    v8f z = {};
    #pragma unroll
    for (int i = 0; i < 4; ++i)
      #pragma unroll
      for (int j = 0; j < 2; ++j) acc[i][j] = z;
  }

  uint4 rAh[2], rAl[2], rBh[2], rBl[2];

  // Preload chunk 0 into buffer 0.
  #pragma unroll
  for (int t = 0; t < 2; ++t) {
    int cidx = tid + t * 256, row = cidx >> 2, g = cidx & 3;   // 128 rows x 4 x 16B
    size_t ia = (size_t)(by * BM + row) * 64 + g;
    size_t ib = (size_t)(bx * BN + row) * 64 + g;
    rAh[t] = pAh[ia]; rAl[t] = pAl[ia];
    rBh[t] = pBh[ib]; rBl[t] = pBl[ib];
  }
  #pragma unroll
  for (int t = 0; t < 2; ++t) {
    int cidx = tid + t * 256, row = cidx >> 2, go = (cidx & 3) << 3;
    *reinterpret_cast<uint4*>(&sAh[0][row * LDA + go]) = rAh[t];
    *reinterpret_cast<uint4*>(&sAl[0][row * LDA + go]) = rAl[t];
    *reinterpret_cast<uint4*>(&sBh[0][row * LDA + go]) = rBh[t];
    *reinterpret_cast<uint4*>(&sBl[0][row * LDA + go]) = rBl[t];
  }
  __syncthreads();

  for (int kc = 0; kc < DD / 32; ++kc) {
    const int p = kc & 1;

    // Issue next chunk's global loads early; latency hides under the WMMA block.
    if (kc + 1 < DD / 32) {
      #pragma unroll
      for (int t = 0; t < 2; ++t) {
        int cidx = tid + t * 256, row = cidx >> 2, g = cidx & 3;
        size_t ia = (size_t)(by * BM + row) * 64 + (size_t)(kc + 1) * 4 + g;
        size_t ib = (size_t)(bx * BN + row) * 64 + (size_t)(kc + 1) * 4 + g;
        rAh[t] = pAh[ia]; rAl[t] = pAl[ia];
        rBh[t] = pBh[ib]; rBl[t] = pBl[ib];
      }
    }

    // 24 WMMAs on buffer p: (hi*hi + hi*lo + lo*hi) split-f16 accumulation.
    v16h bhf[2], blf[2];
    #pragma unroll
    for (int tc = 0; tc < 2; ++tc) {
      const int cb = (wc * 32 + tc * 16) * LDA;
      bhf[tc] = load_frag_b(&sBh[p][cb], lane);
      blf[tc] = load_frag_b(&sBl[p][cb], lane);
    }
    #pragma unroll
    for (int tr = 0; tr < 4; ++tr) {
      const int rb = (wr * 64 + tr * 16) * LDA;
      v16h ahf = load_frag_a(&sAh[p][rb], lane);
      v16h alf = load_frag_a(&sAl[p][rb], lane);
      #pragma unroll
      for (int tc = 0; tc < 2; ++tc) {
        acc[tr][tc] = __builtin_amdgcn_wmma_f32_16x16x32_f16(false, ahf, false, bhf[tc], (short)0, acc[tr][tc], false, false);
        acc[tr][tc] = __builtin_amdgcn_wmma_f32_16x16x32_f16(false, ahf, false, blf[tc], (short)0, acc[tr][tc], false, false);
        acc[tr][tc] = __builtin_amdgcn_wmma_f32_16x16x32_f16(false, alf, false, bhf[tc], (short)0, acc[tr][tc], false, false);
      }
    }

    // Write next chunk into the other buffer; one barrier per iteration.
    if (kc + 1 < DD / 32) {
      #pragma unroll
      for (int t = 0; t < 2; ++t) {
        int cidx = tid + t * 256, row = cidx >> 2, go = (cidx & 3) << 3;
        *reinterpret_cast<uint4*>(&sAh[p ^ 1][row * LDA + go]) = rAh[t];
        *reinterpret_cast<uint4*>(&sAl[p ^ 1][row * LDA + go]) = rAl[t];
        *reinterpret_cast<uint4*>(&sBh[p ^ 1][row * LDA + go]) = rBh[t];
        *reinterpret_cast<uint4*>(&sBl[p ^ 1][row * LDA + go]) = rBl[t];
      }
    }
    __syncthreads();
  }

  // Epilogue: e = exp((dot-1)/T); reduce rows across lanes, cols within lanes.
  float colA0 = 0.f, colA1 = 0.f;
  #pragma unroll
  for (int tr = 0; tr < 4; ++tr) {
    #pragma unroll
    for (int r = 0; r < 8; ++r) {
      float e0 = __expf((acc[tr][0][r] - 1.0f) * INVT);
      float e1 = __expf((acc[tr][1][r] - 1.0f) * INVT);
      colA0 += e0;
      colA1 += e1;
      float rs = e0 + e1;                 // this wave's 32 columns for one row
      rs += __shfl_xor(rs, 1, 32);
      rs += __shfl_xor(rs, 2, 32);
      rs += __shfl_xor(rs, 4, 32);
      rs += __shfl_xor(rs, 8, 32);        // sum over the 16 N-lanes of this half
      if ((lane & 15) == 0)
        rowPart[w][tr * 16 + r + ((lane >> 4) << 3)] = rs;
    }
  }
  colA0 += __shfl_xor(colA0, 16, 32);     // combine M=0..7 half with M=8..15 half
  colA1 += __shfl_xor(colA1, 16, 32);
  if (lane < 16) {
    colPart[w][lane]      = colA0;
    colPart[w][16 + lane] = colA1;
  }
  __syncthreads();

  // Deterministic cross-wave combine, then write block partials to workspace.
  if (tid < 128) {
    int m = tid, mr = m >> 6, ml = m & 63;
    float s = 0.f;
    #pragma unroll
    for (int c = 0; c < 4; ++c) s += rowPart[mr * 4 + c][ml];
    rowG[(size_t)bx * BB + by * BM + m] = s;
  } else {
    int n = tid - 128, nc = n >> 5, nl = n & 31;
    float s = 0.f;
    #pragma unroll
    for (int rr = 0; rr < 2; ++rr) s += colPart[rr * 4 + nc][nl];
    colG[(size_t)by * BVV + bx * BN + n] = s;
  }
}

// ---------------- Kernel 2: positive logits (diagonal), one wave per text ----------------
__global__ __launch_bounds__(256)
void pos_logits(const float* __restrict__ img, const float* __restrict__ txt,
                float* __restrict__ posV) {
  int lane = threadIdx.x & 31;
  int c = blockIdx.x * 8 + (threadIdx.x >> 5);   // 0..20479
  int i = c / VV;
  const float* ip = img + (size_t)i * DD;
  const float* tp = txt + (size_t)c * DD;
  float s = 0.f;
  #pragma unroll
  for (int j = 0; j < DD / 32; ++j) s += ip[lane + j * 32] * tp[lane + j * 32];
  s += __shfl_xor(s, 1, 32);
  s += __shfl_xor(s, 2, 32);
  s += __shfl_xor(s, 4, 32);
  s += __shfl_xor(s, 8, 32);
  s += __shfl_xor(s, 16, 32);
  if (lane == 0) posV[c] = s * INVT;
}

// ---------------- Kernel 3: single-block deterministic final reduction ----------------
__global__ __launch_bounds__(1024)
void final_reduce(const float* __restrict__ rowG, const float* __restrict__ colG,
                  const float* __restrict__ posV, float* __restrict__ out) {
  __shared__ float red[1024];
  int tid = threadIdx.x;
  float aR = 0.f, aC = 0.f, aP = 0.f;
  for (int r = tid; r < BB; r += 1024) {
    float s = 0.f;
    for (int t = 0; t < 160; ++t) s += rowG[(size_t)t * BB + r];
    aR += __logf(s);
  }
  for (int c = tid; c < BVV; c += 1024) {
    float s = 0.f;
    for (int t = 0; t < 32; ++t) s += colG[(size_t)t * BVV + c];
    aC += __logf(s);
    aP += posV[c];
  }
  // loss = 0.5*(mean(logS_row)+SHIFT + mean(logS_col)+SHIFT) - mean(pos)
  red[tid] = aR * (0.5f / BB) + aC * (0.5f / BVV) - aP * (1.0f / BVV);
  __syncthreads();
  for (int s2 = 512; s2 > 0; s2 >>= 1) {
    if (tid < s2) red[tid] += red[tid + s2];
    __syncthreads();
  }
  if (tid == 0) out[0] = red[0] + INVT;
}

extern "C" void kernel_launch(void* const* d_in, const int* in_sizes, int n_in,
                              void* d_out, int out_size, void* d_ws, size_t ws_size,
                              hipStream_t stream) {
  const float* img = (const float*)d_in[0];   // (4096, 512)
  const float* txt = (const float*)d_in[1];   // (4096*5, 512) flattened
  float* rowG = (float*)d_ws;                         // 160 * 4096
  float* colG = rowG + (size_t)160 * BB;              // 32 * 20480
  float* posV = colG + (size_t)32 * BVV;              // 20480
  _Float16* imgH = (_Float16*)(posV + BVV);           // 4096*512
  _Float16* imgL = imgH + (size_t)BB * DD;
  _Float16* txtH = imgL + (size_t)BB * DD;            // 20480*512
  _Float16* txtL = txtH + (size_t)BVV * DD;
  float* out = (float*)d_out;

  const int nImg4 = BB * DD / 4;    // 524288  -> 2048 blocks
  const int nTxt4 = BVV * DD / 4;   // 2621440 -> 10240 blocks
  split_f16<<<nImg4 / 256, 256, 0, stream>>>(img, imgH, imgL, nImg4);
  split_f16<<<nTxt4 / 256, 256, 0, stream>>>(txt, txtH, txtL, nTxt4);
  gemm_lse_partials<<<dim3(BVV / BN, BB / BM), 256, 0, stream>>>(imgH, imgL, txtH, txtL, rowG, colG);
  pos_logits<<<BVV / 8, 256, 0, stream>>>(img, txt, posV);
  final_reduce<<<1, 1024, 0, stream>>>(rowG, colG, posV, out);
}
